// GSGIPCAEMoE_11089605558284
// MI455X (gfx1250) — matrix operand order repaired and checked by reading, not compiled, not run on hardware
//
#include <hip/hip_runtime.h>
#include <hip/hip_bf16.h>

#define N_TOK   65536
#define D_IN    128
#define D_HID   256
#define D_OUT   128
#define N_EXP   8
#define EMBED   64
#define TAU_F   5.0f
#define OUT_ELEMS ((size_t)N_TOK * D_OUT)

typedef __attribute__((ext_vector_type(16))) _Float16 v16h;
typedef __attribute__((ext_vector_type(8)))  _Float16 v8h;
typedef __attribute__((ext_vector_type(8)))  float    v8f;

// ---------------- fragment loaders (CDNA5 WMMA 16x16x32 f16 layouts, ISA 7.12.2) ----
// A 16x32 (MxK) f16: lanes 0-15 row M=lane, K {k0..k0+7, k0+16..k0+23};
//                    lanes 16-31 row M=lane-16, K {k0+8..k0+15, k0+24..k0+31}.
__device__ inline v16h load_frag_a(const _Float16* base, int stride, int m0, int k0, int lane) {
    int l = lane & 15;
    int ksel = (lane >> 4) & 1;
    const _Float16* p = base + (m0 + l) * stride + k0 + ksel * 8;
    union { v16h v; v8h h[2]; } u;
    u.h[0] = *(const v8h*)(p);        // 16B -> ds_load_b128
    u.h[1] = *(const v8h*)(p + 16);
    return u.v;
}
// B 32x16 (KxN) f16 from an n-major (transposed) tile: lanes 0-15 hold K {k0..k0+7, k0+16..k0+23},
// lanes 16-31 hold K {k0+8..k0+15, k0+24..k0+31}; column N = n0 + (lane&15).
__device__ inline v16h load_frag_b(const _Float16* base, int stride, int n0, int k0, int lane) {
    int l = lane & 15;
    int ksel = (lane >> 4) & 1;
    const _Float16* p = base + (n0 + l) * stride + k0 + ksel * 8;
    union { v16h v; v8h h[2]; } u;
    u.h[0] = *(const v8h*)(p);
    u.h[1] = *(const v8h*)(p + 16);
    return u.v;
}

// ---------------- kernel 1: init out / counters / scalar ----------------
__global__ void moe_init(float* __restrict__ out, int* __restrict__ cnt) {
    float4* o4 = (float4*)out;
    size_t total4 = OUT_ELEMS / 4;
    float4 z = make_float4(0.f, 0.f, 0.f, 0.f);
    for (size_t i = (size_t)blockIdx.x * blockDim.x + threadIdx.x; i < total4;
         i += (size_t)gridDim.x * blockDim.x)
        o4[i] = z;
    if (blockIdx.x == 0) {
        if (threadIdx.x < 16) cnt[threadIdx.x] = 0;
        if (threadIdx.x == 16) out[OUT_ELEMS] = 0.0f;  // second tuple output (scalar 0.0)
    }
}

// ---------------- kernel 2: gate + routing lists ----------------
__global__ __launch_bounds__(256) void moe_gate(
    const float* __restrict__ x, const float* __restrict__ Wp,
    const float* __restrict__ bp, const float* __restrict__ E_emb,
    float* __restrict__ p_out, int* __restrict__ cnt,
    int* __restrict__ tokL, float* __restrict__ wgtL)
{
    __shared__ float xs[32][129];               // pad: consecutive tokens hit different banks
    __shared__ float Wps[D_IN * EMBED];
    __shared__ float hs[32][EMBED];
    __shared__ float Es[EMBED * N_EXP];
    __shared__ float ps[32][N_EXP];

    const int tid  = threadIdx.x;
    const int tok0 = blockIdx.x * 32;

    for (int i = tid; i < D_IN * EMBED; i += 256) Wps[i] = Wp[i];
    for (int i = tid; i < EMBED * N_EXP; i += 256) Es[i] = E_emb[i];
    const float4* x4 = (const float4*)(x + (size_t)tok0 * D_IN);
    for (int i = tid; i < 32 * 32; i += 256) {
        int r = i >> 5, c = i & 31;
        float4 v = x4[r * 32 + c];
        xs[r][c*4+0] = v.x; xs[r][c*4+1] = v.y; xs[r][c*4+2] = v.z; xs[r][c*4+3] = v.w;
    }
    __syncthreads();

    // h = x @ Wp + bp  (32 x 64 dots of length 128)
    for (int t = tid; t < 32 * EMBED; t += 256) {
        int r = t >> 6, e = t & 63;
        float acc = bp[e];
        #pragma unroll 8
        for (int k = 0; k < D_IN; ++k) acc += xs[r][k] * Wps[k * EMBED + e];
        hs[r][e] = acc;
    }
    __syncthreads();

    // logits = h @ E_emb ; p = sigmoid(logits / tau)   (one (token,expert) per thread)
    {
        int r = tid >> 3, j = tid & 7;
        float acc = 0.f;
        #pragma unroll 8
        for (int k = 0; k < EMBED; ++k) acc += hs[r][k] * Es[k * N_EXP + j];
        float p = 1.0f / (1.0f + __expf(-acc / TAU_F));
        ps[r][j] = p;
        p_out[(size_t)(tok0 + r) * N_EXP + j] = p;   // third tuple output
    }
    __syncthreads();

    // top-2 (jax semantics: first index wins ties), normalize, append to expert lists
    if (tid < 32) {
        int r = tid;
        int j0 = 0; float v0 = ps[r][0];
        for (int j = 1; j < N_EXP; ++j) { float v = ps[r][j]; if (v > v0) { v0 = v; j0 = j; } }
        int j1 = -1; float v1 = -1.0f;
        for (int j = 0; j < N_EXP; ++j) {
            if (j == j0) continue;
            float v = ps[r][j]; if (v > v1) { v1 = v; j1 = j; }
        }
        float s  = v0 + v1 + 1e-10f;
        float w0 = v0 / s, w1 = v1 / s;
        int n = tok0 + r;
        int p0 = atomicAdd(&cnt[j0], 1);
        tokL[j0 * N_TOK + p0] = n; wgtL[j0 * N_TOK + p0] = w0;
        int p1 = atomicAdd(&cnt[j1], 1);
        tokL[j1 * N_TOK + p1] = n; wgtL[j1 * N_TOK + p1] = w1;
    }
}

// ---------------- kernel 3: routed expert FFN (WMMA f16 -> f32) ----------------
#define XS_STRIDE 144   // 288B rows, 16B aligned
#define HS_STRIDE 272   // 544B rows, 16B aligned
#define EXPERT_LDS_BYTES (64*XS_STRIDE*2 + 128*XS_STRIDE*2 + 64*HS_STRIDE*2 + 256*4 + 128*4 + 64*4 + 64*4)

__global__ __launch_bounds__(256) void moe_expert(
    const float* __restrict__ x,  const float* __restrict__ W1,
    const float* __restrict__ b1, const float* __restrict__ W2,
    const float* __restrict__ b2, const int* __restrict__ cnt,
    const int* __restrict__ tokL, const float* __restrict__ wgtL,
    float* __restrict__ out)
{
    const int TILES = N_TOK / 64;
    const int e    = blockIdx.x / TILES;
    const int tile = blockIdx.x % TILES;
    const int ce   = cnt[e];
    const int row0 = tile * 64;
    if (row0 >= ce) return;                       // uniform across block

    extern __shared__ char smem[];
    _Float16* Xs  = (_Float16*)smem;              // [64][144] gathered x (f16)
    _Float16* Wt  = Xs + 64 * XS_STRIDE;          // [128][144] n-major weight chunk (reused)
    _Float16* Hs  = Wt + 128 * XS_STRIDE;         // [64][272] hidden activations (f16)
    float*    b1s = (float*)(Hs + 64 * HS_STRIDE);// 256
    float*    b2s = b1s + 256;                    // 128
    int*      tkS = (int*)(b2s + 128);            // 64
    float*    wS  = (float*)(tkS + 64);           // 64

    const int tid  = threadIdx.x;
    const int lane = tid & 31;
    const int wv   = tid >> 5;     // 8 waves
    const int mt   = wv & 3;       // M tile: rows mt*16..
    const int ng   = wv >> 2;      // N group: cols ng*64..

    if (tid < 64) {
        int i = row0 + tid;
        if (i < ce) { tkS[tid] = tokL[e * N_TOK + i]; wS[tid] = wgtL[e * N_TOK + i]; }
        else        { tkS[tid] = 0;                   wS[tid] = 0.0f; }
    }
    for (int i = tid; i < D_HID; i += 256) b1s[i] = b1[e * D_HID + i];
    for (int i = tid; i < D_OUT; i += 256) b2s[i] = b2[e * D_OUT + i];
    __syncthreads();

    // gather 64 x rows -> f16 LDS
    const float4* x4 = (const float4*)x;
    for (int i = tid; i < 64 * 32; i += 256) {
        int r = i >> 5, c = i & 31;
        float4 v = x4[(size_t)tkS[r] * (D_IN / 4) + c];
        _Float16* d = Xs + r * XS_STRIDE + c * 4;
        d[0] = (_Float16)v.x; d[1] = (_Float16)v.y; d[2] = (_Float16)v.z; d[3] = (_Float16)v.w;
    }
    __syncthreads();

    const float* W1e = W1 + (size_t)e * D_IN * D_HID;
    const int rbase = mt * 16 + ((lane < 16) ? 0 : 8);
    const int cl = lane & 15;

    // ---- GEMM1: H = relu(Xs[64x128] @ W1[128x256] + b1), 2 n-halves of 128 ----
    for (int nhalf = 0; nhalf < 2; ++nhalf) {
        for (int i = tid; i < 128 * 128; i += 256) {          // Wt[n][k] = W1[k][nh*128+n]
            int k = i >> 7, n = i & 127;
            Wt[n * XS_STRIDE + k] = (_Float16)W1e[k * D_HID + nhalf * 128 + n];
        }
        __syncthreads();
        v8f acc[4];
        #pragma unroll
        for (int j = 0; j < 4; ++j)
            #pragma unroll
            for (int q = 0; q < 8; ++q) acc[j][q] = 0.0f;
        #pragma unroll
        for (int k0 = 0; k0 < 128; k0 += 32) {
            v16h a = load_frag_a(Xs, XS_STRIDE, mt * 16, k0, lane);
            #pragma unroll
            for (int j = 0; j < 4; ++j) {
                v16h b = load_frag_b(Wt, XS_STRIDE, ng * 64 + j * 16, k0, lane);
                acc[j] = __builtin_amdgcn_wmma_f32_16x16x32_f16(
                    false, a, false, b, (short)0, acc[j], false, false);
            }
        }
        #pragma unroll
        for (int j = 0; j < 4; ++j) {                          // bias + ReLU -> f16 Hs
            int n = nhalf * 128 + ng * 64 + j * 16 + cl;
            float bn = b1s[n];
            #pragma unroll
            for (int v = 0; v < 8; ++v) {
                float val = acc[j][v] + bn;
                Hs[(rbase + v) * HS_STRIDE + n] = (_Float16)(val > 0.f ? val : 0.f);
            }
        }
        __syncthreads();
    }

    // ---- GEMM2: OUT = Hs[64x256] @ W2[256x128] + b2, 2 k-halves of 128 ----
    const float* W2e = W2 + (size_t)e * D_HID * D_OUT;
    v8f acc2[4];
    #pragma unroll
    for (int j = 0; j < 4; ++j)
        #pragma unroll
        for (int q = 0; q < 8; ++q) acc2[j][q] = 0.0f;
    for (int khalf = 0; khalf < 2; ++khalf) {
        for (int i = tid; i < 128 * 128; i += 256) {          // Wt[n][kk] = W2[kh*128+kk][n]
            int kk = i >> 7, n = i & 127;
            Wt[n * XS_STRIDE + kk] = (_Float16)W2e[(khalf * 128 + kk) * D_OUT + n];
        }
        __syncthreads();
        #pragma unroll
        for (int k0 = 0; k0 < 128; k0 += 32) {
            v16h a = load_frag_a(Hs, HS_STRIDE, mt * 16, khalf * 128 + k0, lane);
            #pragma unroll
            for (int j = 0; j < 4; ++j) {
                v16h b = load_frag_b(Wt, XS_STRIDE, ng * 64 + j * 16, k0, lane);
                acc2[j] = __builtin_amdgcn_wmma_f32_16x16x32_f16(
                    false, a, false, b, (short)0, acc2[j], false, false);
            }
        }
        __syncthreads();
    }

    // ---- epilogue: (acc + b2) * gate_weight, scatter-accumulate into out ----
    #pragma unroll
    for (int j = 0; j < 4; ++j) {
        int col = ng * 64 + j * 16 + cl;
        float bc = b2s[col];
        #pragma unroll
        for (int v = 0; v < 8; ++v) {
            int r = rbase + v;
            float val = (acc2[j][v] + bc) * wS[r];
            atomicAdd(&out[(size_t)tkS[r] * D_OUT + col], val);
        }
    }
}

// ---------------- launcher ----------------
extern "C" void kernel_launch(void* const* d_in, const int* in_sizes, int n_in,
                              void* d_out, int out_size, void* d_ws, size_t ws_size,
                              hipStream_t stream) {
    const float* x    = (const float*)d_in[0];
    const float* Wp   = (const float*)d_in[1];
    const float* bp   = (const float*)d_in[2];
    const float* Eemb = (const float*)d_in[3];
    const float* W1   = (const float*)d_in[4];
    const float* b1   = (const float*)d_in[5];
    const float* W2   = (const float*)d_in[6];
    const float* b2   = (const float*)d_in[7];
    float* out = (float*)d_out;

    // workspace layout: [16 ints cnt][8*N ints token lists][8*N floats weight lists]
    int*   cnt  = (int*)d_ws;
    int*   tokL = cnt + 16;
    float* wgtL = (float*)(tokL + (size_t)N_EXP * N_TOK);

    float* p_out = out + OUT_ELEMS + 1;   // p_open region of tuple output

    (void)hipFuncSetAttribute((const void*)moe_expert,
                              hipFuncAttributeMaxDynamicSharedMemorySize,
                              EXPERT_LDS_BYTES);

    moe_init<<<2048, 256, 0, stream>>>(out, cnt);
    moe_gate<<<N_TOK / 32, 256, 0, stream>>>(x, Wp, bp, Eemb, p_out, cnt, tokL, wgtL);
    moe_expert<<<N_EXP * (N_TOK / 64), 256, EXPERT_LDS_BYTES, stream>>>(
        x, W1, b1, W2, b2, cnt, tokL, wgtL, out);
}
